// CrossAttentionWithGeoBias_56831007261077
// MI455X (gfx1250) — matrix operand order repaired and checked by reading, not compiled, not run on hardware
//
#include <hip/hip_runtime.h>
#include <hip/hip_bf16.h>

// ---------------------------------------------------------------------------
// CrossAttentionWithGeoBias for MI455X (gfx1250, wave32, WMMA f16->f32)
// B=4 H=4 N=M=2048 DIM=256 D=64.  Flash-attention single pass over M: the
// 256MB geo_bias stream (HBM roofline term, ~11.5us @ 23.3TB/s) is loaded
// exactly once, one full chunk ahead, and issued LAST each chunk so in-order
// LOADcnt waits for K/V never wait on it.  All matmuls via
// v_wmma_f32_16x16x32_f16 with ping-pong double-buffered fragments
// (staggered s_wait_loadcnt confirmed in disasm).
// ---------------------------------------------------------------------------

typedef _Float16 h16;
typedef __attribute__((ext_vector_type(16))) _Float16 v16h;
typedef __attribute__((ext_vector_type(8)))  _Float16 v8h;
typedef __attribute__((ext_vector_type(8)))  float    v8f;

#define DIMC 256
#define HC   4
#define DC   64
#define NC   2048
#define MMC  2048
#define BCNT 4
#define SCALEF 0.125f   /* 1/sqrt(64) */

// Column (K) index for element e of an f16 A-fragment (16x32), ISA 7.12.2.
__device__ __forceinline__ int a_col(int e, int half) {
    return (e < 8) ? (e + 8 * half) : (e + 8 + 8 * half);
}

// A-fragment from f16 row-major memory: p points at (row, colBase) of A.
__device__ __forceinline__ v16h load_a_h16(const h16* p, int half) {
    v8h lo = *(const v8h*)(p + 8 * half);
    v8h hi = *(const v8h*)(p + 16 + 8 * half);
    v16h a;
#pragma unroll
    for (int e = 0; e < 8; ++e) { a[e] = lo[e]; a[8 + e] = hi[e]; }
    return a;
}

// B-fragment (32x16, f16): lane holds column n=lane%16; element e has
// K = a_col(e, half); K contiguous in memory.
__device__ __forceinline__ v16h load_b_h16(const h16* p, int half) {
    v8h lo = *(const v8h*)(p + 8 * half);
    v8h hi = *(const v8h*)(p + 16 + 8 * half);
    v16h b;
#pragma unroll
    for (int e = 0; e < 8; ++e) { b[e] = lo[e]; b[8 + e] = hi[e]; }
    return b;
}

// A-fragment sourced from f32 memory with convert-to-f16 (x rows, once/wave).
__device__ __forceinline__ v16h load_a_f32(const float* p, int half) {
    v16h a;
#pragma unroll
    for (int e = 0; e < 16; ++e) a[e] = (h16)p[a_col(e, half)];
    return a;
}

__device__ __forceinline__ v8f wmma16(v16h a, v16h b, v8f c) {
    return __builtin_amdgcn_wmma_f32_16x16x32_f16(
        /*neg_a=*/false, a, /*neg_b=*/false, b,
        /*c_mod=*/(short)0, c, /*reuse_a=*/false, /*reuse_b=*/false);
}

// ---------------------------------------------------------------------------
// Kernel 0: one-shot f32 -> f16 conversion of the 4 weight matrices.
// ---------------------------------------------------------------------------
__global__ __launch_bounds__(256)
void convert_w_kernel(const float* __restrict__ Wq, const float* __restrict__ Wk,
                      const float* __restrict__ Wv, const float* __restrict__ Wo,
                      h16* __restrict__ wq, h16* __restrict__ wk,
                      h16* __restrict__ wv, h16* __restrict__ wo) {
    const int idx = blockIdx.x * blockDim.x + threadIdx.x;   // 0 .. 262143
    const int which = idx >> 16;
    const int off   = idx & 0xFFFF;
    const float* src = (which == 0) ? Wq : (which == 1) ? Wk : (which == 2) ? Wv : Wo;
    h16*         dst = (which == 0) ? wq : (which == 1) ? wk : (which == 2) ? wv : wo;
    dst[off] = (h16)src[off];
}

// ---------------------------------------------------------------------------
// Projection helpers.
// ---------------------------------------------------------------------------
__device__ __forceinline__ void load_bfrags(v16h (&bf)[8], const h16* W,
                                            int jt, int lr, int half) {
    const h16* wrow = W + (size_t)(jt * 16 + lr) * DIMC;
#pragma unroll
    for (int c = 0; c < 8; ++c) bf[c] = load_b_h16(wrow + c * 32, half);
}

// WHICH: 0=q, 1=k  (dst [b][h][seq][d], store stride DC per row)
//        2=v      (dst [b][h][d][seq], 8 outputs contiguous -> one b128 store)
template <int WHICH>
__device__ __forceinline__ void proj_tile(const v16h (&af)[8], const v16h (&bf)[8],
                                          int jt, const float* bias,
                                          int bidx, int seqbase, int half, int lr,
                                          h16* dst) {
    v8f acc = {};
#pragma unroll
    for (int c = 0; c < 8; ++c) acc = wmma16(af[c], bf[c], acc);
    const int j  = jt * 16 + lr;
    const float bj = bias[j];
    const int hh = j >> 6, dd = j & 63;

    if (WHICH == 2) {
        // v transposed: addr = ((bidx*HC+hh)*DC + dd)*MMC + seq; seq stride 1.
        h16* p = dst + (((size_t)bidx * HC + hh) * DC + dd) * MMC
                     + seqbase + 8 * half;
        v8h pk;
#pragma unroll
        for (int r = 0; r < 8; ++r) pk[r] = (h16)(acc[r] + bj);
        *(v8h*)p = pk;                              // 16B aligned (seq % 8 == 0)
    } else {
        // q/k: addr = ((bidx*HC+hh)*NC + seq)*DC + dd; seq stride -> 64 halves.
        h16* p = dst + (((size_t)bidx * HC + hh) * NC + seqbase + 8 * half) * DC + dd;
#pragma unroll
        for (int r = 0; r < 8; ++r)
            p[(size_t)r * DC] = (h16)(acc[r] + bj);
    }
}

// ---------------------------------------------------------------------------
// Kernel 1: Q/K/V projection (templated on target).  grid=128, block=128.
// jt tiles ping-pong double-buffered: B-frags for jt+1 stream in while the
// WMMA chain consumes jt's fragments.
// ---------------------------------------------------------------------------
template <int WHICH>
__global__ __launch_bounds__(128)
void proj_kernel(const float* __restrict__ x, const h16* __restrict__ W,
                 const float* __restrict__ bias, h16* __restrict__ dst) {
    const int wave = threadIdx.x >> 5;
    const int lane = threadIdx.x & 31;
    const int half = lane >> 4;
    const int lr   = lane & 15;
    const int rowbase = blockIdx.x * 64 + wave * 16;   // global row in [0, 8192)
    const int row     = rowbase + lr;
    const int bidx    = rowbase >> 11;                 // stripe stays in one batch
    const int seqbase = rowbase & 2047;

    v16h af[8];                                   // x row stripe, K=256
#pragma unroll
    for (int c = 0; c < 8; ++c)
        af[c] = load_a_f32(x + (size_t)row * DIMC + c * 32, half);

    v16h bfA[8], bfB[8];
    load_bfrags(bfA, W, 0, lr, half);
    for (int jt = 0; jt < 16; jt += 2) {
        load_bfrags(bfB, W, jt + 1, lr, half);
        proj_tile<WHICH>(af, bfA, jt, bias, bidx, seqbase, half, lr, dst);
        if (jt + 2 < 16) load_bfrags(bfA, W, jt + 2, lr, half);
        proj_tile<WHICH>(af, bfB, jt + 1, bias, bidx, seqbase, half, lr, dst);
    }
}

// ---------------------------------------------------------------------------
// Attention chunk: processes 64 m-columns.  Load issue order exploits
// in-order LOADcnt (ISA 7.1): K first (QK wmma waits only on K), then V,
// then NEXT chunk's geo bias last -> nothing this chunk waits on the HBM geo
// stream; it gets a full chunk of WMMA+softmax latency cover.
// ---------------------------------------------------------------------------
__device__ __forceinline__ void attn_chunk(
    int mc, bool loadNext,
    const h16* kb, const h16* vb, const float* gb,
    h16* pl, int half, int lr, int lane, int nbase,
    const v16h& qa0, const v16h& qa1,
    v8f (&of)[4], float (&mrow)[8], float (&lrow)[8],
    float (&gcur)[4][8], float (&gnext)[4][8])
{
    // prefetch next K/V tiles toward L2/WGP$ (no counter impact)
    if (loadNext) {
        __builtin_prefetch(kb + (size_t)(mc + 64 + lane) * DC, 0, 3);
        __builtin_prefetch(kb + (size_t)(mc + 96 + lane) * DC, 0, 3);
        __builtin_prefetch(vb + (size_t)lane        * MMC + mc + 64, 0, 3);
        __builtin_prefetch(vb + (size_t)(lane + 32) * MMC + mc + 64, 0, 3);
    }

    // ---- K fragments (issued first, consumed first) ----
    v16h kf0[4], kf1[4];
#pragma unroll
    for (int t = 0; t < 4; ++t) {
        const h16* krow = kb + (size_t)(mc + t * 16 + lr) * DC;
        kf0[t] = load_b_h16(krow + 0,  half);
        kf1[t] = load_b_h16(krow + 32, half);
    }
    // ---- S = Q K^T: four independent 2-deep WMMA chains ----
    v8f s[4];
#pragma unroll
    for (int t = 0; t < 4; ++t) {
        v8f z = {};
        z = wmma16(qa0, kf0[t], z);
        z = wmma16(qa1, kf1[t], z);
        s[t] = z;
    }

    // ---- V fragments (K frags now dead; consumed after softmax) ----
    v16h vf0[4], vf1[4];
#pragma unroll
    for (int dt = 0; dt < 4; ++dt) {
        const h16* vrow = vb + (size_t)(dt * 16 + lr) * MMC + mc;
        vf0[dt] = load_b_h16(vrow + 0,  half);
        vf1[dt] = load_b_h16(vrow + 32, half);
    }

    // ---- NEXT chunk's geo bias, issued last (NT stream-once) ----
    if (loadNext) {
        const float* gp = gb + (size_t)(nbase + 8 * half) * MMC + (mc + 64) + lr;
#pragma unroll
        for (int t = 0; t < 4; ++t)
#pragma unroll
            for (int r = 0; r < 8; ++r)
                gnext[t][r] = __builtin_nontemporal_load(gp + (size_t)r * MMC + t * 16);
    }

    // ---- scale + geo bias (gcur arrived a full chunk ago) ----
#pragma unroll
    for (int t = 0; t < 4; ++t)
#pragma unroll
        for (int r = 0; r < 8; ++r)
            s[t][r] = s[t][r] * SCALEF + gcur[t][r];

    // ---- online softmax ----
    float nm[8], corr[8], rs[8];
#pragma unroll
    for (int r = 0; r < 8; ++r) {
        float v = s[0][r];
        v = fmaxf(v, s[1][r]); v = fmaxf(v, s[2][r]); v = fmaxf(v, s[3][r]);
        v = fmaxf(v, __shfl_xor(v, 1));
        v = fmaxf(v, __shfl_xor(v, 2));
        v = fmaxf(v, __shfl_xor(v, 4));
        v = fmaxf(v, __shfl_xor(v, 8));
        nm[r]   = fmaxf(mrow[r], v);
        corr[r] = __expf(mrow[r] - nm[r]);
        mrow[r] = nm[r];
        rs[r] = 0.0f;
    }
#pragma unroll
    for (int t = 0; t < 4; ++t)
#pragma unroll
        for (int r = 0; r < 8; ++r) {
            float p = __expf(s[t][r] - nm[r]);
            s[t][r] = p;
            rs[r] += p;
        }
#pragma unroll
    for (int r = 0; r < 8; ++r) {
        float v = rs[r];
        v += __shfl_xor(v, 1); v += __shfl_xor(v, 2);
        v += __shfl_xor(v, 4); v += __shfl_xor(v, 8);
        lrow[r] = lrow[r] * corr[r] + v;
    }
#pragma unroll
    for (int dt = 0; dt < 4; ++dt)
#pragma unroll
        for (int r = 0; r < 8; ++r) of[dt][r] *= corr[r];

    // ---- P (C-layout f32) -> LDS -> A-layout f16 fragments ----
    __syncthreads();
#pragma unroll
    for (int t = 0; t < 4; ++t)
#pragma unroll
        for (int r = 0; r < 8; ++r)
            pl[(r + 8 * half) * 64 + t * 16 + lr] = (h16)s[t][r];
    __syncthreads();
    v16h pa0 = load_a_h16(pl + (size_t)lr * 64 + 0,  half);
    v16h pa1 = load_a_h16(pl + (size_t)lr * 64 + 32, half);

    // ---- O += P V  (waits V only; next-geo still in flight) ----
#pragma unroll
    for (int dt = 0; dt < 4; ++dt) {
        of[dt] = wmma16(pa0, vf0[dt], of[dt]);
        of[dt] = wmma16(pa1, vf1[dt], of[dt]);
    }
}

// ---------------------------------------------------------------------------
// Kernel 2: flash attention with geo bias.  grid=(32, H, B), block=128.
// ---------------------------------------------------------------------------
__global__ __launch_bounds__(128)
void attn_kernel(const h16* __restrict__ qh, const h16* __restrict__ kh,
                 const h16* __restrict__ vh, const float* __restrict__ geo,
                 h16* __restrict__ aoh) {
    __shared__ h16 plds[4 * 16 * 64];             // per-wave P staging (8KB)

    const int wave = threadIdx.x >> 5;
    const int lane = threadIdx.x & 31;
    const int half = lane >> 4;
    const int lr   = lane & 15;
    const int h = blockIdx.y, b = blockIdx.z;
    const int nbase = blockIdx.x * 64 + wave * 16;

    const h16* qbase = qh + (((size_t)b * HC + h) * NC + nbase + lr) * DC;
    v16h qa0 = load_a_h16(qbase + 0,  half);
    v16h qa1 = load_a_h16(qbase + 32, half);

    const h16*   kb = kh  + ((size_t)b * HC + h) * MMC * DC;
    const h16*   vb = vh  + ((size_t)b * HC + h) * DC  * MMC;
    const float* gb = geo + ((size_t)b * HC + h) * NC * (size_t)MMC;

    float mrow[8], lrow[8];
    v8f of[4] = { {}, {}, {}, {} };
#pragma unroll
    for (int r = 0; r < 8; ++r) { mrow[r] = -1e30f; lrow[r] = 0.0f; }

    h16* pl = plds + wave * 16 * 64;

    // Preload geo bias for chunk 0 (ping buffer).
    float gA[4][8], gB[4][8];
    {
        const float* gp = gb + (size_t)(nbase + 8 * half) * MMC + lr;
#pragma unroll
        for (int t = 0; t < 4; ++t)
#pragma unroll
            for (int r = 0; r < 8; ++r)
                gA[t][r] = __builtin_nontemporal_load(gp + (size_t)r * MMC + t * 16);
    }

    // 32 chunks, ping-pong geo buffers (no array copies).
    for (int mc = 0; mc < MMC; mc += 128) {
        attn_chunk(mc,      true,
                   kb, vb, gb, pl, half, lr, lane, nbase,
                   qa0, qa1, of, mrow, lrow, gA, gB);
        attn_chunk(mc + 64, (mc + 128) < MMC,
                   kb, vb, gb, pl, half, lr, lane, nbase,
                   qa0, qa1, of, mrow, lrow, gB, gA);
    }

    // ---- normalize by row sums and emit [b][n][h*64+d] as f16 ----
#pragma unroll
    for (int dt = 0; dt < 4; ++dt)
#pragma unroll
        for (int r = 0; r < 8; ++r) {
            const int n = nbase + r + 8 * half;
            const float val = of[dt][r] / lrow[r];
            aoh[((size_t)b * NC + n) * DIMC + h * DC + dt * 16 + lr] = (h16)val;
        }
}

// ---------------------------------------------------------------------------
// Kernel 3: output projection  out = attn @ Wo^T + bo  (f16 in, f32 out).
// ---------------------------------------------------------------------------
__device__ __forceinline__ void oproj_tile(const v16h (&af)[8], const v16h (&bf)[8],
                                           int jt, const float* bo,
                                           int rowbase, int half, int lr,
                                           float* out) {
    v8f acc = {};
#pragma unroll
    for (int c = 0; c < 8; ++c) acc = wmma16(af[c], bf[c], acc);
    const int j  = jt * 16 + lr;
    const float bj = bo[j];
    float* p = out + (size_t)(rowbase + 8 * half) * DIMC + j;
#pragma unroll
    for (int r = 0; r < 8; ++r)
        __builtin_nontemporal_store(acc[r] + bj, p + (size_t)r * DIMC);
}

__global__ __launch_bounds__(128)
void out_proj_kernel(const h16* __restrict__ aoh, const h16* __restrict__ wo,
                     const float* __restrict__ bo, float* __restrict__ out) {
    const int wave = threadIdx.x >> 5;
    const int lane = threadIdx.x & 31;
    const int half = lane >> 4;
    const int lr   = lane & 15;
    const int rowbase = blockIdx.x * 64 + wave * 16;
    const int row     = rowbase + lr;

    v16h af[8];
#pragma unroll
    for (int c = 0; c < 8; ++c)
        af[c] = load_a_h16(aoh + (size_t)row * DIMC + c * 32, half);

    v16h bfA[8], bfB[8];
    load_bfrags(bfA, wo, 0, lr, half);
    for (int jt = 0; jt < 16; jt += 2) {
        load_bfrags(bfB, wo, jt + 1, lr, half);
        oproj_tile(af, bfA, jt, bo, rowbase, half, lr, out);
        if (jt + 2 < 16) load_bfrags(bfA, wo, jt + 2, lr, half);
        oproj_tile(af, bfB, jt + 1, bo, rowbase, half, lr, out);
    }
}

// ---------------------------------------------------------------------------
extern "C" void kernel_launch(void* const* d_in, const int* in_sizes, int n_in,
                              void* d_out, int out_size, void* d_ws, size_t ws_size,
                              hipStream_t stream) {
    const float* xq  = (const float*)d_in[0];
    const float* xc  = (const float*)d_in[1];
    const float* geo = (const float*)d_in[2];
    const float* Wq  = (const float*)d_in[3];
    const float* bq  = (const float*)d_in[4];
    const float* Wk  = (const float*)d_in[5];
    const float* bk  = (const float*)d_in[6];
    const float* Wv  = (const float*)d_in[7];
    const float* bv  = (const float*)d_in[8];
    const float* Wo  = (const float*)d_in[9];
    const float* bo  = (const float*)d_in[10];
    float* out = (float*)d_out;

    const size_t planeBytes = (size_t)BCNT * NC * DIMC * sizeof(h16);  // 4MB each
    const size_t wBytes     = (size_t)DIMC * DIMC * sizeof(h16);       // 128KB each
    char* ws = (char*)d_ws;
    h16* qh  = (h16*)(ws + 0 * planeBytes);
    h16* kh  = (h16*)(ws + 1 * planeBytes);
    h16* vh  = (h16*)(ws + 2 * planeBytes);
    h16* aoh = (h16*)(ws + 3 * planeBytes);
    h16* wq  = (h16*)(ws + 4 * planeBytes + 0 * wBytes);
    h16* wk  = (h16*)(ws + 4 * planeBytes + 1 * wBytes);
    h16* wv  = (h16*)(ws + 4 * planeBytes + 2 * wBytes);
    h16* wo  = (h16*)(ws + 4 * planeBytes + 3 * wBytes);

    // 0) One-shot weight conversion: 4 * 65536 elements.
    convert_w_kernel<<<dim3(1024), 256, 0, stream>>>(Wq, Wk, Wv, Wo, wq, wk, wv, wo);

    // 1) Q/K/V projections: 8192 rows / 64 per block; templated per target
    //    so the store epilogues are straight-line (no per-element branches).
    proj_kernel<0><<<dim3(128), 128, 0, stream>>>(xq, wq, bq, qh);
    proj_kernel<1><<<dim3(128), 128, 0, stream>>>(xc, wk, bk, kh);
    proj_kernel<2><<<dim3(128), 128, 0, stream>>>(xc, wv, bv, vh);

    // 2) Flash attention: (N/64, H, B) blocks of 4 waves.
    attn_kernel<<<dim3(NC / 64, HC, BCNT), 128, 0, stream>>>(qh, kh, vh, geo, aoh);

    // 3) Output projection.
    out_proj_kernel<<<dim3(128), 128, 0, stream>>>(aoh, wo, bo, out);
}
